// MLAttention_32298154066586
// MI455X (gfx1250) — compile-verified, matrix operand order checked by reading, not run on hardware
//
#include <hip/hip_runtime.h>
#include <hip/hip_bf16.h>

// ---------------------------------------------------------------------------
// MLA (DeepSeek-style) for MI455X / gfx1250, wave32, WMMA bf16 path.
// All GEMMs + attention einsums run on v_wmma_f32_16x16x32_bf16.
// GEMM tiles are double-buffered via CDNA5 async memory->LDS DMA (ASYNCcnt),
// overlapping DMA with WMMA using partial s_wait_asynccnt.
// ---------------------------------------------------------------------------

typedef unsigned short u16;
typedef __attribute__((ext_vector_type(16))) __bf16 v16bf;
typedef __attribute__((ext_vector_type(8)))  float  v8f;

#define SEQ    2048
#define DMODEL 2048
#define NH     16
#define KVL    512
#define RP     64
#define QKD    576           // KVL + RP
#define QLORA  1536
#define HQK    (NH * QKD)    // 9216
#define HKV    (NH * KVL)    // 8192

__device__ __forceinline__ u16 f2bf(float f) {
  unsigned u = __float_as_uint(f);
  unsigned r = u + 0x7FFFu + ((u >> 16) & 1u);   // round-to-nearest-even
  return (u16)(r >> 16);
}

// CDNA5 async global->LDS copy (GLOBAL_LOAD_ASYNC_TO_LDS_B128, ASYNCcnt).
// LDS operand is the workgroup-relative byte address = low 32 bits of the
// generic pointer to a __shared__ object.
__device__ __forceinline__ void async_copy_b128(void* lds, const void* gptr) {
  unsigned loff = (unsigned)(uintptr_t)lds;
  asm volatile("global_load_async_to_lds_b128 %0, %1, off"
               :: "v"(loff), "v"(gptr) : "memory");
}
__device__ __forceinline__ void async_wait0() {
  asm volatile("s_wait_asynccnt 0x0" ::: "memory");
}
// Wait until at most N async ops are outstanding (in-order completion =>
// everything older than the newest N has landed in LDS).
template <int N>
__device__ __forceinline__ void async_wait_le() {
  asm volatile("s_wait_asynccnt %0" :: "i"(N) : "memory");
}

// A/B fragment load for V_WMMA_F32_16X16X32_BF16.
// Element (row, k) lives at base[row*stride + k]; K-pairs are contiguous.
// Layout per ISA 7.12.2: row = lane%16; VGPR r holds
//   K = (r/4)*16 + (lane/16)*8 + (r%4)*2 .. +1
__device__ __forceinline__ v16bf load_frag(const u16* base, int stride, int lane) {
  int row = lane & 15, half = lane >> 4;
  union { unsigned u[8]; v16bf v; } f;
#pragma unroll
  for (int r = 0; r < 8; ++r) {
    int kb = (r >> 2) * 16 + half * 8 + (r & 3) * 2;
    f.u[r] = *(const unsigned*)(base + (size_t)row * stride + kb);
  }
  return f.v;
}

__device__ __forceinline__ v8f wmma_bf16(v16bf a, v16bf b, v8f c) {
  return __builtin_amdgcn_wmma_f32_16x16x32_bf16(false, a, false, b, (short)0, c, false, false);
}

// ---------------------------------------------------------------------------
// 1) f32 -> bf16 convert
// ---------------------------------------------------------------------------
__global__ __launch_bounds__(256) void cvt_f32_bf16(const float* __restrict__ x,
                                                    u16* __restrict__ y, int n) {
  for (int i = blockIdx.x * 256 + threadIdx.x; i < n; i += gridDim.x * 256)
    y[i] = f2bf(x[i]);
}

// ---------------------------------------------------------------------------
// 2) Generic GEMM: C[m][n] = sum_k A[m][k] * W[n][k]   (A,W bf16, C f32)
//    BM=128, BN in {128, 64}, BK=32. 256 threads = 8 waves in a 4x2 grid;
//    each wave owns a 32 x (BN/2) tile -> 2 x (BN/32) WMMA subtiles.
//    LDS tiles double-buffered; async DMA overlapped with WMMA.
// ---------------------------------------------------------------------------
#define BM 128
#define BK 32
#define LDAS 40   // padded LDS strides (bank-conflict avoidance)

template <int BN>
__device__ __forceinline__ void stage_tiles(const u16* __restrict__ A,
                                            const u16* __restrict__ W,
                                            u16* As, u16* Bs,
                                            int bm, int bn, int K, int k0, int tid) {
#pragma unroll
  for (int i = 0; i < (BM * BK / 8) / 256; ++i) {
    int idx = tid + i * 256;
    int row = idx >> 2, ch = idx & 3;
    async_copy_b128(As + row * LDAS + ch * 8,
                    A + (size_t)(bm + row) * K + k0 + ch * 8);
  }
#pragma unroll
  for (int i = 0; i < (BN * BK / 8) / 256; ++i) {
    int idx = tid + i * 256;
    int row = idx >> 2, ch = idx & 3;
    async_copy_b128(Bs + row * LDAS + ch * 8,
                    W + (size_t)(bn + row) * K + k0 + ch * 8);
  }
}

template <int BN>
__global__ __launch_bounds__(256) void gemm_bf16_nt(const u16* __restrict__ A,
                                                    const u16* __restrict__ W,
                                                    float* __restrict__ C,
                                                    int M, int N, int K) {
  constexpr int WN = BN / 2;    // wave n extent (64 or 32)
  constexpr int TN = WN / 16;   // n-subtiles per wave (4 or 2)
  constexpr int PER = (BM * BK / 8) / 256 + (BN * BK / 8) / 256;  // async ops/tile/wave
  __shared__ u16 As[2][BM * LDAS];
  __shared__ u16 Bs[2][BN * LDAS];
  const int tid = threadIdx.x, lane = tid & 31, wave = tid >> 5;
  const int bm = blockIdx.y * BM, bn = blockIdx.x * BN;
  const int wm = (wave >> 1) * 32, wn = (wave & 1) * WN;
  const int NK = K / BK;

  v8f c[2][TN] = {};

  stage_tiles<BN>(A, W, As[0], Bs[0], bm, bn, K, 0, tid);

  for (int kt = 0; kt < NK; ++kt) {
    const int cur = kt & 1;
    if (kt + 1 < NK) {
      // prefetch next tile into the other buffer, then wait only for the
      // current tile (in-order ASYNCcnt => <= PER means older batch landed)
      stage_tiles<BN>(A, W, As[cur ^ 1], Bs[cur ^ 1], bm, bn, K, (kt + 1) * BK, tid);
      async_wait_le<PER>();
    } else {
      async_wait0();
    }
    __syncthreads();

    v16bf a[2], b[TN];
#pragma unroll
    for (int i = 0; i < 2; ++i)
      a[i] = load_frag(As[cur] + (wm + i * 16) * LDAS, LDAS, lane);
#pragma unroll
    for (int j = 0; j < TN; ++j)
      b[j] = load_frag(Bs[cur] + (wn + j * 16) * LDAS, LDAS, lane);
#pragma unroll
    for (int i = 0; i < 2; ++i)
#pragma unroll
      for (int j = 0; j < TN; ++j)
        c[i][j] = wmma_bf16(a[i], b[j], c[i][j]);

    __syncthreads();   // all waves done reading buf[cur] before it is restaged
  }

  const int n = lane & 15, half = lane >> 4;
#pragma unroll
  for (int i = 0; i < 2; ++i)
#pragma unroll
    for (int j = 0; j < TN; ++j)
#pragma unroll
      for (int r = 0; r < 8; ++r) {
        int m = r + 8 * half;
        C[(size_t)(bm + wm + i * 16 + m) * N + (bn + wn + j * 16 + n)] = c[i][j][r];
      }
}

// ---------------------------------------------------------------------------
// 3) RMS-norm rows (full width), f32 in -> bf16 out. One block per row.
// ---------------------------------------------------------------------------
__global__ __launch_bounds__(256) void rmsnorm_rows(const float* __restrict__ X,
                                                    const float* __restrict__ w,
                                                    u16* __restrict__ Y, int ncols) {
  const int row = blockIdx.x, tid = threadIdx.x;
  const float* x = X + (size_t)row * ncols;
  float ss = 0.f;
  for (int c = tid; c < ncols; c += 256) { float v = x[c]; ss += v * v; }
  for (int off = 16; off; off >>= 1) ss += __shfl_xor(ss, off, 32);
  __shared__ float red[8];
  __shared__ float rinv;
  if ((tid & 31) == 0) red[tid >> 5] = ss;
  __syncthreads();
  if (tid == 0) {
    float tot = 0.f;
    for (int i = 0; i < 8; ++i) tot += red[i];
    rinv = rsqrtf(tot / (float)ncols + 1e-6f);
  }
  __syncthreads();
  for (int c = tid; c < ncols; c += 256)
    Y[(size_t)row * ncols + c] = f2bf(w[c] * x[c] * rinv);
}

// ---------------------------------------------------------------------------
// 4) kv build: ckv f32 [S,576] -> kv bf16 [S,576] (RMS on first 512, RoPE on
//    last 64) and V^T bf16 [512, S] for the P@V WMMA B-fragments.
// ---------------------------------------------------------------------------
__global__ __launch_bounds__(256) void kv_build(const float* __restrict__ ckv,
                                                const float* __restrict__ wln,
                                                const float* __restrict__ cosb,
                                                const float* __restrict__ sinb,
                                                u16* __restrict__ kv,
                                                u16* __restrict__ vT) {
  const int s = blockIdx.x, tid = threadIdx.x;
  const float* x = ckv + (size_t)s * QKD;
  float ss = 0.f;
  for (int c = tid; c < KVL; c += 256) { float v = x[c]; ss += v * v; }
  for (int off = 16; off; off >>= 1) ss += __shfl_xor(ss, off, 32);
  __shared__ float red[8];
  __shared__ float rinv;
  if ((tid & 31) == 0) red[tid >> 5] = ss;
  __syncthreads();
  if (tid == 0) {
    float tot = 0.f;
    for (int i = 0; i < 8; ++i) tot += red[i];
    rinv = rsqrtf(tot / (float)KVL + 1e-6f);
  }
  __syncthreads();
  for (int c = tid; c < KVL; c += 256) {
    u16 v = f2bf(wln[c] * x[c] * rinv);
    kv[(size_t)s * QKD + c] = v;
    vT[(size_t)c * SEQ + s] = v;
  }
  if (tid < RP) {
    int j = tid;
    float xx = x[KVL + j];
    float x2 = x[KVL + (j < 32 ? j + 32 : j - 32)];
    float rh = (j < 32) ? -x2 : x2;
    kv[(size_t)s * QKD + KVL + j] =
        f2bf(xx * cosb[(size_t)s * RP + j] + rh * sinb[(size_t)s * RP + j]);
  }
}

// ---------------------------------------------------------------------------
// 5) q RoPE (per head, last 64 of each 576 block) + f32 -> bf16
// ---------------------------------------------------------------------------
__global__ __launch_bounds__(256) void q_rope_cvt(const float* __restrict__ qf,
                                                  const float* __restrict__ cosb,
                                                  const float* __restrict__ sinb,
                                                  u16* __restrict__ qb) {
  const int s = blockIdx.x, tid = threadIdx.x;
  const float* qr = qf + (size_t)s * HQK;
  for (int i = tid; i < HQK; i += 256) {
    int h = i / QKD, off = i % QKD;
    float v;
    if (off < KVL) {
      v = qr[i];
    } else {
      int j = off - KVL;
      float xx = qr[h * QKD + KVL + j];
      float x2 = qr[h * QKD + KVL + (j < 32 ? j + 32 : j - 32)];
      float rh = (j < 32) ? -x2 : x2;
      v = xx * cosb[(size_t)s * RP + j] + rh * sinb[(size_t)s * RP + j];
    }
    qb[(size_t)s * HQK + i] = f2bf(v);
  }
}

// ---------------------------------------------------------------------------
// 6) MLA attention. One block = (16 queries) x (1 head). 256 threads, 8 waves.
//    Full 16x2048 f32 score tile in LDS (no online softmax; 320KB/WGP LDS).
//    kv (2.4 MB bf16) streams from L2 as B-fragments.
// ---------------------------------------------------------------------------
#define SQ_STR 584     // 576 + pad
#define SP_STR 2056    // 2048 + pad
#define SMEM_SS   (16 * SEQ * 4)              // 131072
#define SMEM_SP   (16 * SP_STR * 2)           // 65792
#define SMEM_SQ   (16 * SQ_STR * 2)           // 18688
#define SMEM_MASK (16 * 64 * 4)               // 4096
#define SMEM_ATTN (SMEM_SS + SMEM_SP + SMEM_SQ + SMEM_MASK + 64)

__global__ __launch_bounds__(256) void mla_attn(const u16* __restrict__ Q,
                                                const u16* __restrict__ KV,
                                                const u16* __restrict__ VT,
                                                const int* __restrict__ topk,
                                                u16* __restrict__ O) {
  extern __shared__ char smem[];
  float*    sS    = (float*)smem;                                    // [16][2048]
  u16*      sP    = (u16*)(smem + SMEM_SS);                          // [16][SP_STR]
  u16*      sQ    = (u16*)(smem + SMEM_SS + SMEM_SP);                // [16][SQ_STR]
  unsigned* sMask = (unsigned*)(smem + SMEM_SS + SMEM_SP + SMEM_SQ); // [16][64]
  float*    sSum  = (float*)(smem + SMEM_SS + SMEM_SP + SMEM_SQ + SMEM_MASK);

  const int tid = threadIdx.x, lane = tid & 31, wave = tid >> 5;
  const int s0 = blockIdx.x * 16, h = blockIdx.y;
  const float scale = 1.0f / 24.0f;   // QK^-0.5 = 1/sqrt(576)

  // ---- stage q tile (16 x 576 bf16 = 16 x 72 chunks of 16B) async ----
  for (int i = tid; i < 16 * (QKD / 8); i += 256) {
    int row = i / (QKD / 8), ch = i % (QKD / 8);
    async_copy_b128(sQ + row * SQ_STR + ch * 8,
                    Q + (size_t)(s0 + row) * HQK + h * QKD + ch * 8);
  }
  // ---- topk ∧ causal bitmask ----
  for (int i = tid; i < 16 * 64; i += 256) sMask[i] = 0u;
  __syncthreads();
  for (int i = tid; i < 16 * 512; i += 256) {
    int row = i >> 9, j = i & 511;
    int t = topk[(size_t)(s0 + row) * 512 + j];
    atomicOr(&sMask[row * 64 + (t >> 5)], 1u << (t & 31));
  }
  async_wait0();
  __syncthreads();

  // ---- pass 1: scores = q · kv^T  (per wave: 16 key tiles of 16) ----
  for (int nt = wave; nt < SEQ / 16; nt += 8) {
    v8f acc = {};
#pragma unroll
    for (int kk = 0; kk < QKD; kk += 32) {
      v16bf a = load_frag(sQ + kk, SQ_STR, lane);
      v16bf b = load_frag(KV + (size_t)nt * 16 * QKD + kk, QKD, lane);
      acc = wmma_bf16(a, b, acc);
    }
    int n = lane & 15, half = lane >> 4;
    int t = nt * 16 + n;
#pragma unroll
    for (int r = 0; r < 8; ++r) {
      int m = r + 8 * half;
      unsigned bit = (sMask[m * 64 + (t >> 5)] >> (t & 31)) & 1u;
      float v = acc[r] * scale;
      if (!(bit && t <= s0 + m)) v = -__builtin_inff();
      sS[m * SEQ + t] = v;
    }
  }
  __syncthreads();

  // ---- softmax (each wave: 2 rows; shuffle reductions; 1/sum deferred) ----
#pragma unroll
  for (int rr = 0; rr < 2; ++rr) {
    int m = wave * 2 + rr;
    float mx = -__builtin_inff();
    for (int c = lane; c < SEQ; c += 32) mx = fmaxf(mx, sS[m * SEQ + c]);
    for (int off = 16; off; off >>= 1) mx = fmaxf(mx, __shfl_xor(mx, off, 32));
    float sum = 0.f;
    for (int c = lane; c < SEQ; c += 32) {
      float e = __expf(sS[m * SEQ + c] - mx);
      sum += e;
      sP[m * SP_STR + c] = f2bf(e);
    }
    for (int off = 16; off; off >>= 1) sum += __shfl_xor(sum, off, 32);
    if (lane == 0) sSum[m] = sum;
  }
  __syncthreads();

  // ---- pass 2: O = P · V  (per wave: 4 d-tiles of 16; K = 2048) ----
#pragma unroll
  for (int j = 0; j < 4; ++j) {
    int nd = wave * 4 + j;
    v8f acc = {};
    for (int kk = 0; kk < SEQ; kk += 32) {
      v16bf a = load_frag(sP + kk, SP_STR, lane);
      v16bf b = load_frag(VT + (size_t)nd * 16 * SEQ + kk, SEQ, lane);
      acc = wmma_bf16(a, b, acc);
    }
    int n = lane & 15, half = lane >> 4;
#pragma unroll
    for (int r = 0; r < 8; ++r) {
      int m = r + 8 * half;
      float v = acc[r] / sSum[m];
      O[(size_t)(s0 + m) * HKV + h * KVL + nd * 16 + n] = f2bf(v);
    }
  }
}

// ---------------------------------------------------------------------------
// Host: pipeline launch
// ---------------------------------------------------------------------------
extern "C" void kernel_launch(void* const* d_in, const int* in_sizes, int n_in,
                              void* d_out, int out_size, void* d_ws, size_t ws_size,
                              hipStream_t stream) {
  const float* hidden = (const float*)d_in[0];   // [S, D]
  const float* cosb   = (const float*)d_in[1];   // [S, 64]
  const float* sinb   = (const float*)d_in[2];   // [S, 64]
  const int*   topk   = (const int*)d_in[3];     // [S, 512]
  const float* w_q_a  = (const float*)d_in[4];   // [1536, 2048]
  const float* w_q_a_ln = (const float*)d_in[5]; // [1536]
  const float* w_q_b  = (const float*)d_in[6];   // [9216, 1536]
  const float* w_kv_a = (const float*)d_in[7];   // [576, 2048]
  const float* w_kv_a_ln = (const float*)d_in[8];// [512]
  const float* w_o    = (const float*)d_in[9];   // [2048, 8192]
  float* out = (float*)d_out;

  char* ws = (char*)d_ws;
  size_t cur = 0;
  auto alloc = [&](size_t bytes) { char* p = ws + cur; cur += (bytes + 255) & ~size_t(255); return p; };

  u16*   hidden_bf = (u16*)alloc((size_t)SEQ * DMODEL * 2);
  u16*   wqa_bf    = (u16*)alloc((size_t)QLORA * DMODEL * 2);
  u16*   wqb_bf    = (u16*)alloc((size_t)HQK * QLORA * 2);
  u16*   wkva_bf   = (u16*)alloc((size_t)QKD * DMODEL * 2);
  u16*   wo_bf     = (u16*)alloc((size_t)DMODEL * HKV * 2);
  float* qa_f      = (float*)alloc((size_t)SEQ * QLORA * 4);
  u16*   qresid_bf = (u16*)alloc((size_t)SEQ * QLORA * 2);
  float* ckv_f     = (float*)alloc((size_t)SEQ * QKD * 4);
  u16*   kv_bf     = (u16*)alloc((size_t)SEQ * QKD * 2);
  u16*   vT_bf     = (u16*)alloc((size_t)KVL * SEQ * 2);
  float* q_f       = (float*)alloc((size_t)SEQ * HQK * 4);
  u16*   q_bf      = (u16*)alloc((size_t)SEQ * HQK * 2);
  u16*   attn_bf   = (u16*)alloc((size_t)SEQ * HKV * 2);

  // 1) convert inputs/weights to bf16
  cvt_f32_bf16<<<1024, 256, 0, stream>>>(hidden, hidden_bf, SEQ * DMODEL);
  cvt_f32_bf16<<<1024, 256, 0, stream>>>(w_q_a, wqa_bf, QLORA * DMODEL);
  cvt_f32_bf16<<<1024, 256, 0, stream>>>(w_q_b, wqb_bf, HQK * QLORA);
  cvt_f32_bf16<<<1024, 256, 0, stream>>>(w_kv_a, wkva_bf, QKD * DMODEL);
  cvt_f32_bf16<<<1024, 256, 0, stream>>>(w_o, wo_bf, DMODEL * HKV);

  // 2) q_a = hidden @ w_q_a.T ; ckv = hidden @ w_kv_a.T
  gemm_bf16_nt<128><<<dim3(QLORA / 128, SEQ / BM), 256, 0, stream>>>(
      hidden_bf, wqa_bf, qa_f, SEQ, QLORA, DMODEL);
  gemm_bf16_nt<64><<<dim3(QKD / 64, SEQ / BM), 256, 0, stream>>>(
      hidden_bf, wkva_bf, ckv_f, SEQ, QKD, DMODEL);

  // 3) RMS norms + kv/rope build
  rmsnorm_rows<<<SEQ, 256, 0, stream>>>(qa_f, w_q_a_ln, qresid_bf, QLORA);
  kv_build<<<SEQ, 256, 0, stream>>>(ckv_f, w_kv_a_ln, cosb, sinb, kv_bf, vT_bf);

  // 4) q = q_resid @ w_q_b.T ; then RoPE on pe slice + cvt
  gemm_bf16_nt<128><<<dim3(HQK / 128, SEQ / BM), 256, 0, stream>>>(
      qresid_bf, wqb_bf, q_f, SEQ, HQK, QLORA);
  q_rope_cvt<<<SEQ, 256, 0, stream>>>(q_f, cosb, sinb, q_bf);

  // 5) attention (dynamic LDS ≈ 215 KB; CDNA5 allows up to 320 KB per WG)
  (void)hipFuncSetAttribute((const void*)mla_attn,
                            hipFuncAttributeMaxDynamicSharedMemorySize, SMEM_ATTN);
  mla_attn<<<dim3(SEQ / 16, NH), 256, SMEM_ATTN, stream>>>(
      q_bf, kv_bf, vT_bf, topk, attn_bf);

  // 6) out = attn @ w_o.T  (f32 straight to d_out)
  gemm_bf16_nt<128><<<dim3(DMODEL / 128, SEQ / BM), 256, 0, stream>>>(
      attn_bf, wo_bf, out, SEQ, DMODEL, HKV);

  (void)in_sizes; (void)n_in; (void)out_size; (void)ws_size;
}
